// PointTransformerBlock_55817394979224
// MI455X (gfx1250) — matrix operand order repaired and checked by reading, not compiled, not run on hardware
//
#include <hip/hip_runtime.h>

typedef __attribute__((ext_vector_type(16))) _Float16 v16h;
typedef __attribute__((ext_vector_type(8)))  float    v8f;

// ---------------- problem constants ----------------
constexpr int S_    = 4096;   // seq len / points
constexpr int NB    = 8;      // batch
constexpr int DIM   = 128;    // d_model
constexpr int MOUT  = 2048;   // FPS output points
constexpr int KNNK  = 8;      // knn neighbors
constexpr int HID   = 512;    // ffn hidden
constexpr int PK    = 160;    // padded 3+DIM=131 -> 160 (mult of 32)
constexpr int PN    = 256;    // pool width 2*DIM
constexpr float QSCALE = 0.08838834764831845f; // 1/sqrt(128)

// ---------------- workspace layout (bytes, overlapped) ----------------
constexpr size_t OFF_H32   = 0;                         // 16 MiB f32 [S*B,128]
constexpr size_t OFF_HFIN  = OFF_H32  + 16777216;       // 16 MiB f32 [B,N,128]
constexpr size_t OFF_H16   = OFF_HFIN + 16777216;       // 8 MiB  f16 [B*S,128]
constexpr size_t OFF_Q16   = OFF_H16  + 8388608;        // 8 MiB
constexpr size_t OFF_K16   = OFF_Q16  + 8388608;        // 8 MiB
constexpr size_t OFF_VT16  = OFF_K16  + 8388608;        // 8 MiB f16 [DIM][B*S] (V^T)
constexpr size_t OFF_VOUT  = OFF_VT16 + 8388608;        // 8 MiB
constexpr size_t OFF_ATTN  = OFF_VOUT + 8388608;        // 16 MiB f32 (reused: hp2 f32)
constexpr size_t OFF_FFIN  = OFF_ATTN + 16777216;       // 8 MiB f16 (reused: pooled f16)
constexpr size_t OFF_FF1   = OFF_FFIN + 8388608;        // 32 MiB f16 [32768,512]
constexpr size_t OFF_FF2   = OFF_FF1  + 33554432;       // 16 MiB f32
constexpr size_t OFF_APOOL = OFF_H16;                   // reuse: 40 MiB f16 [131072,160]
constexpr size_t OFF_HP1   = OFF_FF1;                   // reuse: 64 MiB f16 [131072,256]
constexpr size_t OFF_W     = OFF_HP1 + 67108864;        // swizzled weights (f16)
constexpr size_t OFF_WQ    = OFF_W;                     // 4 x 32 KiB
constexpr size_t OFF_WK    = OFF_WQ + 32768;
constexpr size_t OFF_WV    = OFF_WK + 32768;
constexpr size_t OFF_WO    = OFF_WV + 32768;
constexpr size_t OFF_W1    = OFF_WO + 32768;            // 128x512
constexpr size_t OFF_W2    = OFF_W1 + 131072;           // 512x128
constexpr size_t OFF_WPIN  = OFF_W2 + 131072;           // 160x256 padded
constexpr size_t OFF_WPOUT = OFF_WPIN + 81920;          // 256x256
constexpr size_t OFF_KNN   = OFF_WPOUT + 131072;        // int [B*M*K]

// ---------------- WMMA fragment helpers (CDNA5 16-bit layouts) ----------------
__device__ __forceinline__ v8f wmma32(v16h a, v16h b, v8f c) {
  return __builtin_amdgcn_wmma_f32_16x16x32_f16(false, a, false, b, (short)0, c,
                                                false, false);
}

// A fragment 16x32 (MxK), row-major source (one contiguous 32B run per lane).
__device__ __forceinline__ v16h load_a16(const _Float16* Arow, int lda, int kc) {
  union { v16h v; unsigned u[8]; } f;
  int lane = threadIdx.x & 31;
  int m = lane & 15, hi = lane >> 4;
  const _Float16* r = Arow + (size_t)m * lda + kc;
#pragma unroll
  for (int p = 0; p < 8; ++p) {
    int k0 = hi * 16 + 2 * p;
    f.u[p] = *(const unsigned*)(r + k0);
  }
  return f.v;
}

// B fragment from pre-swizzled weights: lane's 16 halves contiguous (32B load).
__device__ __forceinline__ v16h load_b16_swz(const _Float16* W, int ntiles,
                                             int kc, int nc) {
  int lane = threadIdx.x & 31;
  const _Float16* p =
      W + (((size_t)(kc >> 5) * ntiles + (nc >> 4)) * 32 + lane) * 16;
  return *(const v16h*)p;
}

// B fragment where B[k][n] = Src[(r0+n)*ld + c0 + k]  (k contiguous)
__device__ __forceinline__ v16h load_bT16(const _Float16* Src, int ld, int r0, int c0) {
  union { v16h v; unsigned u[8]; } f;
  int lane = threadIdx.x & 31;
  int n = lane & 15, hi = (lane >> 4) & 1;
  const _Float16* r = Src + (size_t)(r0 + n) * ld + c0 + hi * 16;
#pragma unroll
  for (int p = 0; p < 8; ++p) f.u[p] = *(const unsigned*)(r + 2 * p);
  return f.v;
}

__device__ __forceinline__ float red16_max(float v) {
  v = fmaxf(v, __shfl_xor(v, 1, 32));
  v = fmaxf(v, __shfl_xor(v, 2, 32));
  v = fmaxf(v, __shfl_xor(v, 4, 32));
  v = fmaxf(v, __shfl_xor(v, 8, 32));
  return v;
}
__device__ __forceinline__ float red16_sum(float v) {
  v += __shfl_xor(v, 1, 32);
  v += __shfl_xor(v, 2, 32);
  v += __shfl_xor(v, 4, 32);
  v += __shfl_xor(v, 8, 32);
  return v;
}

// ---------------- CDNA5 async copy global -> LDS (ASYNCcnt tracked) ----------
// Generic pointers to __shared__ carry the LDS byte offset in their low 32 bits
// which is what VDST wants.
__device__ __forceinline__ void async_b128(void* lds, const void* g) {
  unsigned loff = (unsigned)(size_t)lds;
  asm volatile("global_load_async_to_lds_b128 %0, %1, off"
               :: "v"(loff), "v"(g) : "memory");
}
__device__ __forceinline__ void wait_async0() {
  asm volatile("s_wait_asynccnt 0" ::: "memory");
}

// stage one 32-key block: K tile (32 rows x 128 halves, globally contiguous)
// and V^T tile (128 rows x 32 halves, strided rows). 128 threads, 8 x b128 each.
__device__ __forceinline__ void stage_tiles(const _Float16* Kb, const _Float16* VTb,
                                            int t0, _Float16* dK, _Float16* dV,
                                            int tid) {
#pragma unroll
  for (int j = 0; j < 4; ++j) {
    int c = tid + j * 128;                       // 512 chunks of 8 halves
    async_b128(dK + c * 8, Kb + (size_t)t0 * DIM + c * 8);
  }
#pragma unroll
  for (int j = 0; j < 4; ++j) {
    int c = tid + j * 128;
    int row = c >> 2, part = c & 3;
    async_b128(dV + row * 32 + part * 8,
               VTb + (size_t)row * (NB * S_) + t0 + part * 8);
  }
}

// ---------------- weight swizzle: f32 [K,N] -> fragment-order f16 -------------
__global__ void swz_weight_k(const float* __restrict__ src, _Float16* __restrict__ dst,
                             int K, int N, int srcK) {
  int i = blockIdx.x * 256 + threadIdx.x;
  if (i >= K * N) return;
  int tile = i >> 9;
  int rem = i & 511;
  int lane = rem >> 4;
  int idx = rem & 15;
  int ntiles = N >> 4;
  int tk = tile / ntiles, tn = tile - tk * ntiles;
  int k = tk * 32 + (lane >> 4) * 16 + idx;
  int n = tn * 16 + (lane & 15);
  float v = (k < srcK) ? src[(size_t)k * N + n] : 0.0f;
  dst[i] = (_Float16)v;
}

// ---------------- rmsnorm rows of 128 ----------------------------------------
__global__ void rmsnorm_k(const float* __restrict__ X, const float* __restrict__ scale,
                          float* __restrict__ out32, _Float16* __restrict__ out16,
                          int permuteBS) {
  int r = blockIdx.x;
  int t = threadIdx.x;          // 128
  float v = X[(size_t)r * DIM + t];
  float sq = v * v;
#pragma unroll
  for (int o = 16; o > 0; o >>= 1) sq += __shfl_xor(sq, o, 32);
  __shared__ float ssum[4];
  if ((t & 31) == 0) ssum[t >> 5] = sq;
  __syncthreads();
  float tot = ssum[0] + ssum[1] + ssum[2] + ssum[3];
  float rms = sqrtf(tot) * 0.08838834764831845f;   // norm/sqrt(D)
  float y = scale[t] * (v / (rms + 1e-8f));
  if (out32) out32[(size_t)r * DIM + t] = y;
  size_t orow = r;
  if (permuteBS) { int s = r >> 3, b = r & 7; orow = (size_t)b * S_ + s; }
  out16[orow * DIM + t] = (_Float16)y;
}

// ---------------- WMMA GEMM, 16 rows x 64 cols per wave -----------------------
__global__ void __launch_bounds__(32)
gemm_wmma_k(const _Float16* __restrict__ A, int lda,
            const _Float16* __restrict__ Bswz, int ntiles,
            const float* __restrict__ bias, float scale, int leaky,
            _Float16* __restrict__ C16, int c16_rs, int c16_cs,
            float* __restrict__ C32, int ldc32, int Kdim) {
  int tn0 = blockIdx.x * 4;                    // 4 n-tiles = 64 cols
  int tm = blockIdx.y;
  const _Float16* Arow = A + (size_t)tm * 16 * lda;
  v8f acc[4] = {};
  for (int kc = 0; kc < Kdim; kc += 32) {
    v16h a  = load_a16(Arow, lda, kc);
    v16h b0 = load_b16_swz(Bswz, ntiles, kc, (tn0 + 0) * 16);
    v16h b1 = load_b16_swz(Bswz, ntiles, kc, (tn0 + 1) * 16);
    v16h b2 = load_b16_swz(Bswz, ntiles, kc, (tn0 + 2) * 16);
    v16h b3 = load_b16_swz(Bswz, ntiles, kc, (tn0 + 3) * 16);
    acc[0] = wmma32(a, b0, acc[0]);
    acc[1] = wmma32(a, b1, acc[1]);
    acc[2] = wmma32(a, b2, acc[2]);
    acc[3] = wmma32(a, b3, acc[3]);
  }
  int lane = threadIdx.x & 31;
  int n = lane & 15, hi = lane >> 4;
#pragma unroll
  for (int t = 0; t < 4; ++t) {
    int col = (tn0 + t) * 16 + n;
    float bv = bias ? bias[col] : 0.0f;
#pragma unroll
    for (int p = 0; p < 8; ++p) {
      int row = tm * 16 + hi * 8 + p;
      float v = (acc[t][p] + bv) * scale;
      if (leaky) v = v > 0.0f ? v : 0.01f * v;
      if (C32) C32[(size_t)row * ldc32 + col] = v;
      if (C16) C16[(size_t)row * c16_rs + (size_t)col * c16_cs] = (_Float16)v;
    }
  }
}

// ---------------- flash attention: 4 waves/WG share async-staged K/V tiles ----
__global__ void __launch_bounds__(128)
flash_attn_k(const _Float16* __restrict__ Q, const _Float16* __restrict__ Km,
             const _Float16* __restrict__ VT, _Float16* __restrict__ O) {
  int b = blockIdx.y;
  int tid = threadIdx.x;
  int wid = tid >> 5;                       // wave 0..3
  int qt = blockIdx.x * 4 + wid;            // this wave's 16-query tile
  const _Float16* Qb = Q + (size_t)b * S_ * DIM;
  const _Float16* Kb = Km + (size_t)b * S_ * DIM;
  const _Float16* VTb = VT + (size_t)b * S_;     // row stride NB*S_
  int lane = tid & 31;
  int n = lane & 15, hi = (lane >> 4) & 1;

  __shared__ _Float16 sK[2][32 * DIM];      // 2 x 8 KiB, rows = key
  __shared__ _Float16 sV[2][DIM * 32];      // 2 x 8 KiB, rows = d
  __shared__ _Float16 sP[4][16 * 32];       // per-wave P tile

  v16h qa[4];
#pragma unroll
  for (int c = 0; c < 4; ++c)
    qa[c] = load_a16(Qb + (size_t)qt * 16 * DIM, DIM, c * 32);

  v8f vz = {};
  v8f acc[8];
#pragma unroll
  for (int i = 0; i < 8; ++i) acc[i] = vz;
  float mrow[8], lrow[8];
#pragma unroll
  for (int p = 0; p < 8; ++p) { mrow[p] = -INFINITY; lrow[p] = 0.0f; }

  // prologue: stage tile 0
  stage_tiles(Kb, VTb, 0, sK[0], sV[0], tid);
  wait_async0();
  __syncthreads();

  for (int t0 = 0; t0 < S_; t0 += 32) {
    int cur = (t0 >> 5) & 1;
    if (t0 + 32 < S_)                       // stream next tile into other buffer
      stage_tiles(Kb, VTb, t0 + 32, sK[cur ^ 1], sV[cur ^ 1], tid);

    const _Float16* cK = sK[cur];
    const _Float16* cV = sV[cur];

    // S = Q K^T for 32 keys (from LDS, k-contiguous -> ds_load_b128)
    v8f s0 = vz, s1 = vz;
#pragma unroll
    for (int c = 0; c < 4; ++c) {
      s0 = wmma32(qa[c], load_bT16(cK, DIM, 0, c * 32), s0);
      s1 = wmma32(qa[c], load_bT16(cK, DIM, 16, c * 32), s1);
    }
    // online softmax (q carries 1/sqrt(D))
    float corr[8];
#pragma unroll
    for (int p = 0; p < 8; ++p) {
      float mb = red16_max(fmaxf(s0[p], s1[p]));
      float mnew = fmaxf(mrow[p], mb);
      float c = __expf(mrow[p] - mnew);
      float p0 = __expf(s0[p] - mnew);
      float p1 = __expf(s1[p] - mnew);
      lrow[p] = lrow[p] * c + red16_sum(p0 + p1);
      mrow[p] = mnew;
      corr[p] = c;
      s0[p] = p0;
      s1[p] = p1;
    }
#pragma unroll
    for (int dt = 0; dt < 8; ++dt)
#pragma unroll
      for (int p = 0; p < 8; ++p) acc[dt][p] = acc[dt][p] * corr[p];

    // transpose P (C-layout -> A-layout) through this wave's LDS slice
    _Float16* myP = sP[wid];
#pragma unroll
    for (int p = 0; p < 8; ++p) {
      myP[(hi * 8 + p) * 32 + n]      = (_Float16)s0[p];
      myP[(hi * 8 + p) * 32 + 16 + n] = (_Float16)s1[p];
    }
    asm volatile("s_wait_dscnt 0" ::: "memory");   // intra-wave RAW on sP
    v16h pa = load_a16(myP, 32, 0);

    // acc += P @ V : B[k][n] = cV[(dt*16+n)*32 + k]
#pragma unroll
    for (int dt = 0; dt < 8; ++dt)
      acc[dt] = wmma32(pa, load_bT16(cV, 32, dt * 16, 0), acc[dt]);

    wait_async0();                          // my share of next tile landed
    __syncthreads();                        // everyone's share landed
  }

#pragma unroll
  for (int dt = 0; dt < 8; ++dt)
#pragma unroll
    for (int p = 0; p < 8; ++p) {
      int row = qt * 16 + hi * 8 + p;
      O[((size_t)b * S_ + row) * DIM + dt * 16 + n] = (_Float16)(acc[dt][p] / lrow[p]);
    }
}

// ---------------- residual + relayout: hfin[b][n][d] = h32[n*B+b] + ff2[b*S+n]
__global__ void residual_k(const float* __restrict__ h32, const float* __restrict__ ff2,
                           float* __restrict__ hfin) {
  size_t i = (size_t)blockIdx.x * 256 + threadIdx.x;   // over B*N*D
  int d = i & 127;
  size_t row = i >> 7;                                 // b*N + n
  int nn = row & (S_ - 1);
  int b = (int)(row >> 12);
  hfin[i] = h32[((size_t)nn * NB + b) * DIM + d] + ff2[i];
}

// ---------------- farthest point sampling: one block per batch ---------------
__global__ void fps_k(const float* __restrict__ P, float* __restrict__ fp) {
  int b = blockIdx.x;
  const float* Pb = P + (size_t)b * S_ * 3;
  float* fpb = fp + (size_t)b * MOUT * 3;
  __shared__ float slast[3];
  __shared__ float sred[256];
  __shared__ int   sidx[256];
  int tid = threadIdx.x;
  float dloc[S_ / 256];
#pragma unroll
  for (int i = 0; i < S_ / 256; ++i) dloc[i] = INFINITY;
  int last = 0;
  if (tid == 0) { fpb[0] = Pb[0]; fpb[1] = Pb[1]; fpb[2] = Pb[2]; }
  for (int it = 1; it < MOUT; ++it) {
    if (tid == 0) {
      slast[0] = Pb[last * 3];
      slast[1] = Pb[last * 3 + 1];
      slast[2] = Pb[last * 3 + 2];
    }
    __syncthreads();
    float lx = slast[0], ly = slast[1], lz = slast[2];
    float best = -1.0f;
    int besti = 0;
#pragma unroll
    for (int i = 0; i < S_ / 256; ++i) {
      int nn = i * 256 + tid;
      float dx = Pb[nn * 3] - lx, dy = Pb[nn * 3 + 1] - ly, dz = Pb[nn * 3 + 2] - lz;
      float d = fminf(dloc[i], dx * dx + dy * dy + dz * dz);
      dloc[i] = d;
      if (d > best) { best = d; besti = nn; }
    }
    sred[tid] = best;
    sidx[tid] = besti;
    __syncthreads();
    for (int st = 128; st > 0; st >>= 1) {
      if (tid < st) {
        if (sred[tid + st] > sred[tid] ||
            (sred[tid + st] == sred[tid] && sidx[tid + st] < sidx[tid])) {
          sred[tid] = sred[tid + st];
          sidx[tid] = sidx[tid + st];
        }
      }
      __syncthreads();
    }
    last = sidx[0];
    if (tid == 0) {
      fpb[it * 3]     = Pb[last * 3];
      fpb[it * 3 + 1] = Pb[last * 3 + 1];
      fpb[it * 3 + 2] = Pb[last * 3 + 2];
    }
    __syncthreads();
  }
}

// ---------------- KNN: one thread per (b,m), insertion top-8 -----------------
__global__ void knn_k(const float* __restrict__ P, const float* __restrict__ fp,
                      int* __restrict__ knn) {
  int q = blockIdx.x * 128 + threadIdx.x;
  if (q >= NB * MOUT) return;
  int b = q / MOUT;
  float fx = fp[(size_t)q * 3], fy = fp[(size_t)q * 3 + 1], fz = fp[(size_t)q * 3 + 2];
  const float* Pb = P + (size_t)b * S_ * 3;
  float bd[KNNK];
  int bi[KNNK];
#pragma unroll
  for (int j = 0; j < KNNK; ++j) { bd[j] = INFINITY; bi[j] = 0; }
  for (int nn = 0; nn < S_; ++nn) {
    float dx = Pb[nn * 3] - fx, dy = Pb[nn * 3 + 1] - fy, dz = Pb[nn * 3 + 2] - fz;
    float d = dx * dx + dy * dy + dz * dz;
    if (d < bd[KNNK - 1]) {
      int j = KNNK - 1;
      while (j > 0 && bd[j - 1] > d) { bd[j] = bd[j - 1]; bi[j] = bi[j - 1]; --j; }
      bd[j] = d;
      bi[j] = nn;
    }
  }
#pragma unroll
  for (int j = 0; j < KNNK; ++j) knn[(size_t)q * KNNK + j] = bi[j];
}

// ---------------- build pool-MLP A matrix: [B*M*K, 160] f16 ------------------
__global__ void gather_k(const float* __restrict__ fp, const int* __restrict__ knn,
                         const float* __restrict__ hfin, _Float16* __restrict__ Ap) {
  int row = blockIdx.x;          // b*M*K + m*K + k
  int c = threadIdx.x;           // 160
  int bm = row >> 3;
  float v;
  if (c < 3) {
    v = fp[(size_t)bm * 3 + c];
  } else if (c < 131) {
    int nn = knn[row];
    int b = bm / MOUT;
    v = hfin[((size_t)b * S_ + nn) * DIM + (c - 3)];
  } else {
    v = 0.0f;
  }
  Ap[(size_t)row * PK + c] = (_Float16)v;
}

// ---------------- max-pool over K neighbors ----------------------------------
__global__ void maxpool_k(const _Float16* __restrict__ hp1, _Float16* __restrict__ pooled) {
  int bm = blockIdx.x;
  int c = threadIdx.x;           // 256
  float mx = -INFINITY;
#pragma unroll
  for (int k = 0; k < KNNK; ++k)
    mx = fmaxf(mx, (float)hp1[((size_t)bm * KNNK + k) * PN + c]);
  pooled[(size_t)bm * PN + c] = (_Float16)mx;
}

// ---------------- [b*M+m, c] -> out[m, b, c] ---------------------------------
__global__ void out_relayout_k(const float* __restrict__ hp2, float* __restrict__ out) {
  size_t i = (size_t)blockIdx.x * 256 + threadIdx.x;   // over B*M*256
  int c = i & 255;
  size_t r = i >> 8;
  int b = (int)(r / MOUT);
  int m = (int)(r % MOUT);
  out[((size_t)m * NB + b) * PN + c] = hp2[i];
}

// =============================== launcher ====================================
extern "C" void kernel_launch(void* const* d_in, const int* in_sizes, int n_in,
                              void* d_out, int out_size, void* d_ws, size_t ws_size,
                              hipStream_t stream) {
  const float* x          = (const float*)d_in[0];
  const float* p_pos      = (const float*)d_in[1];
  const float* scale_attn = (const float*)d_in[2];
  const float* scale_ff   = (const float*)d_in[3];
  const float* Wq = (const float*)d_in[4];  const float* bq = (const float*)d_in[5];
  const float* Wk = (const float*)d_in[6];  const float* bk = (const float*)d_in[7];
  const float* Wv = (const float*)d_in[8];  const float* bv = (const float*)d_in[9];
  const float* Wo = (const float*)d_in[10]; const float* bo = (const float*)d_in[11];
  const float* W1 = (const float*)d_in[12]; const float* b1 = (const float*)d_in[13];
  const float* W2 = (const float*)d_in[14]; const float* b2 = (const float*)d_in[15];
  const float* Wpin  = (const float*)d_in[16];
  const float* Wpout = (const float*)d_in[17];

  char* w = (char*)d_ws;
  float*     h32   = (float*)(w + OFF_H32);
  float*     hfin  = (float*)(w + OFF_HFIN);
  _Float16*  h16   = (_Float16*)(w + OFF_H16);
  _Float16*  q16   = (_Float16*)(w + OFF_Q16);
  _Float16*  k16   = (_Float16*)(w + OFF_K16);
  _Float16*  vt16  = (_Float16*)(w + OFF_VT16);
  _Float16*  vout  = (_Float16*)(w + OFF_VOUT);
  float*     attn  = (float*)(w + OFF_ATTN);
  _Float16*  ffin  = (_Float16*)(w + OFF_FFIN);
  _Float16*  ff1   = (_Float16*)(w + OFF_FF1);
  float*     ff2   = (float*)(w + OFF_FF2);
  _Float16*  Ap    = (_Float16*)(w + OFF_APOOL);
  _Float16*  hp1   = (_Float16*)(w + OFF_HP1);
  float*     hp2   = (float*)(w + OFF_ATTN);     // reuse
  _Float16*  pooled= (_Float16*)(w + OFF_FFIN);  // reuse
  _Float16*  wq16  = (_Float16*)(w + OFF_WQ);
  _Float16*  wk16  = (_Float16*)(w + OFF_WK);
  _Float16*  wv16  = (_Float16*)(w + OFF_WV);
  _Float16*  wo16  = (_Float16*)(w + OFF_WO);
  _Float16*  w116  = (_Float16*)(w + OFF_W1);
  _Float16*  w216  = (_Float16*)(w + OFF_W2);
  _Float16*  wpin16  = (_Float16*)(w + OFF_WPIN);
  _Float16*  wpout16 = (_Float16*)(w + OFF_WPOUT);
  int*       knn   = (int*)(w + OFF_KNN);

  float* out_hp = (float*)d_out;
  float* out_fp = out_hp + (size_t)MOUT * NB * PN;   // fp_pos region of output

  dim3 b32(32), b128(128), b256(256);

  // ---- weight swizzle f32 -> fragment-order f16 ----
  swz_weight_k<<<64, b256, 0, stream>>>(Wq, wq16, DIM, DIM, DIM);
  swz_weight_k<<<64, b256, 0, stream>>>(Wk, wk16, DIM, DIM, DIM);
  swz_weight_k<<<64, b256, 0, stream>>>(Wv, wv16, DIM, DIM, DIM);
  swz_weight_k<<<64, b256, 0, stream>>>(Wo, wo16, DIM, DIM, DIM);
  swz_weight_k<<<256, b256, 0, stream>>>(W1, w116, DIM, HID, DIM);
  swz_weight_k<<<256, b256, 0, stream>>>(W2, w216, HID, DIM, HID);
  swz_weight_k<<<160, b256, 0, stream>>>(Wpin, wpin16, PK, PN, 131);
  swz_weight_k<<<256, b256, 0, stream>>>(Wpout, wpout16, PN, PN, PN);

  // ---- h = rmsnorm(x, scale_attn); f32 in [s*B+b] order, f16 in [b*S+s] ----
  rmsnorm_k<<<S_ * NB, b128, 0, stream>>>(x, scale_attn, h32, h16, 1);

  // ---- q/k/v projections (q pre-scaled; V written transposed) ----
  gemm_wmma_k<<<dim3(DIM / 64, (NB * S_) / 16), b32, 0, stream>>>(
      h16, DIM, wq16, DIM / 16, bq, QSCALE, 0, q16, DIM, 1, nullptr, 0, DIM);
  gemm_wmma_k<<<dim3(DIM / 64, (NB * S_) / 16), b32, 0, stream>>>(
      h16, DIM, wk16, DIM / 16, bk, 1.0f, 0, k16, DIM, 1, nullptr, 0, DIM);
  gemm_wmma_k<<<dim3(DIM / 64, (NB * S_) / 16), b32, 0, stream>>>(
      h16, DIM, wv16, DIM / 16, bv, 1.0f, 0, vt16, 1, NB * S_, nullptr, 0, DIM);

  // ---- flash attention (4 query tiles per WG, async double-buffered K/V) ----
  flash_attn_k<<<dim3(S_ / 64, NB), b128, 0, stream>>>(q16, k16, vt16, vout);

  // ---- O projection ----
  gemm_wmma_k<<<dim3(DIM / 64, (NB * S_) / 16), b32, 0, stream>>>(
      vout, DIM, wo16, DIM / 16, bo, 1.0f, 0, nullptr, 0, 0, attn, DIM, DIM);

  // ---- FFN: rmsnorm -> W1(leaky) -> W2 ----
  rmsnorm_k<<<S_ * NB, b128, 0, stream>>>(attn, scale_ff, nullptr, ffin, 0);
  gemm_wmma_k<<<dim3(HID / 64, (NB * S_) / 16), b32, 0, stream>>>(
      ffin, DIM, w116, HID / 16, b1, 1.0f, 1, ff1, HID, 1, nullptr, 0, DIM);
  gemm_wmma_k<<<dim3(DIM / 64, (NB * S_) / 16), b32, 0, stream>>>(
      ff1, HID, w216, DIM / 16, b2, 1.0f, 0, nullptr, 0, 0, ff2, DIM, HID);

  // ---- residual + relayout into [B,N,D] ----
  residual_k<<<(NB * S_ * DIM) / 256, b256, 0, stream>>>(h32, ff2, hfin);

  // ---- FPS (writes fp_pos straight into output) ----
  fps_k<<<NB, b256, 0, stream>>>(p_pos, out_fp);

  // ---- KNN ----
  knn_k<<<(NB * MOUT) / 128, b128, 0, stream>>>(p_pos, out_fp, knn);

  // ---- gather concat(fp_pos, knn features) -> padded f16 A matrix ----
  gather_k<<<NB * MOUT * KNNK, dim3(PK), 0, stream>>>(out_fp, knn, hfin, Ap);

  // ---- pool MLP: leaky(A @ Wpin) then max over K, then @ Wpout ----
  gemm_wmma_k<<<dim3(PN / 64, (NB * MOUT * KNNK) / 16), b32, 0, stream>>>(
      Ap, PK, wpin16, PN / 16, nullptr, 1.0f, 1, hp1, PN, 1, nullptr, 0, PK);
  maxpool_k<<<NB * MOUT, b256, 0, stream>>>(hp1, pooled);
  gemm_wmma_k<<<dim3(PN / 64, (NB * MOUT) / 16), b32, 0, stream>>>(
      pooled, PN, wpout16, PN / 16, nullptr, 1.0f, 0, nullptr, 0, 0, hp2, PN, PN);

  // ---- relayout [b*M+m, c] -> [m, b, c] into output ----
  out_relayout_k<<<(NB * MOUT * PN) / 256, b256, 0, stream>>>(hp2, out_hp);
}
